// MultiHeadAttention_23545010716930
// MI455X (gfx1250) — matrix operand order repaired and checked by reading, not compiled, not run on hardware
//
#include <hip/hip_runtime.h>
#include <hip/hip_bf16.h>
#include <math.h>

// Problem constants (from reference setup_inputs)
#define Bsz 4
#define Tt  2048
#define Cc  2048
#define Hh  16
#define Dd  128
#define Mm  (Bsz * Tt)     // 8192
#define N_QKV (3 * Cc)     // 6144

typedef __attribute__((ext_vector_type(16))) __bf16 v16bf;
typedef __attribute__((ext_vector_type(8)))  float  v8f;

union FragBF { v16bf v; uint4 q[2]; __bf16 h[16]; };
union AccU   { v8f   v; float f[8]; };

#define LDT2 72   // LDS tile row stride in bf16 elements (64 + 8 pad); 144B, 16B-aligned

// ---- CDNA5 async global->LDS copy (ASYNCcnt-tracked, bypasses VGPRs) -------
static __device__ __forceinline__ void async_copy_b128(const void* gsrc, void* ldst) {
  // Generic LDS pointer: addr[31:0] is the LDS offset (aperture in high bits).
  unsigned lds = (unsigned)(size_t)ldst;
  unsigned long long g = (unsigned long long)(size_t)gsrc;
  asm volatile("global_load_async_to_lds_b128 %0, %1, off"
               :: "v"(lds), "v"(g) : "memory");
}

#if defined(__has_builtin)
# if __has_builtin(__builtin_amdgcn_s_wait_asynccnt)
#  define WAIT_ASYNC(N) __builtin_amdgcn_s_wait_asynccnt(N)
# endif
#endif
#ifndef WAIT_ASYNC
# define WAIT_ASYNC(N) asm volatile("s_wait_asynccnt %0" :: "i"(N) : "memory")
#endif

static __device__ __forceinline__ v8f wmma_bf16(v16bf a, v16bf b, v8f c) {
  // 8 args: (neg_a, A, neg_b, B, c_mod, C, reuse_a, reuse_b)
  return __builtin_amdgcn_wmma_f32_16x16x32_bf16(false, a, false, b, (short)0, c,
                                                 false, false);
}

// A operand fragment (16x32 bf16, MxK) from row-major memory.
// Layout: lane holds row M=lane&15; elems 0..15 at k = half*16 + e
// (contiguous, matching B fragments)  -> two contiguous 16B chunks.
static __device__ __forceinline__ v16bf load_a_frag(const __bf16* base, int row0,
                                                    int stride, int lane) {
  const int half = lane >> 4, m = lane & 15;
  const __bf16* p = base + (size_t)(row0 + m) * stride;
  FragBF f;
  f.q[0] = *(const uint4*)(p + half * 16);
  f.q[1] = *(const uint4*)(p + half * 16 + 8);
  return f.v;
}

// B operand fragment (32x16 bf16, KxN); memory holds B^T row-major so lane
// (col N = lane&15) reads 16 contiguous k values starting at half*16.
static __device__ __forceinline__ v16bf load_b_frag(const __bf16* base, int nrow0,
                                                    int stride, int lane) {
  const int half = lane >> 4, n = lane & 15;
  const __bf16* p = base + (size_t)(nrow0 + n) * stride + half * 16;
  FragBF f;
  f.q[0] = ((const uint4*)p)[0];
  f.q[1] = ((const uint4*)p)[1];
  return f.v;
}

static __device__ __forceinline__ v16bf load_b_row(const __bf16* p) {
  FragBF f;
  f.q[0] = ((const uint4*)p)[0];
  f.q[1] = ((const uint4*)p)[1];
  return f.v;
}

static __device__ __forceinline__ float red_max16(float v) {
  v = fmaxf(v, __shfl_xor(v, 1, 32));
  v = fmaxf(v, __shfl_xor(v, 2, 32));
  v = fmaxf(v, __shfl_xor(v, 4, 32));
  v = fmaxf(v, __shfl_xor(v, 8, 32));
  return v;
}
static __device__ __forceinline__ float red_sum16(float v) {
  v += __shfl_xor(v, 1, 32);
  v += __shfl_xor(v, 2, 32);
  v += __shfl_xor(v, 4, 32);
  v += __shfl_xor(v, 8, 32);
  return v;
}

// ---------------------------------------------------------------------------
// One-shot fp32 -> bf16 downcast (so GEMM tiles can use async LDS copies and
// halve their global read traffic). ~192MiB streamed @ 23.3TB/s: negligible.
// ---------------------------------------------------------------------------
__global__ __launch_bounds__(256) void cvt_f32_to_bf16(const float* __restrict__ src,
                                                       __bf16* __restrict__ dst,
                                                       int n) {
  const int i = (blockIdx.x * 256 + threadIdx.x) * 8;
  if (i >= n) return;
  float4 f0 = ((const float4*)(src + i))[0];
  float4 f1 = ((const float4*)(src + i))[1];
  union { uint4 q; __bf16 h[8]; } u;
  u.h[0] = (__bf16)f0.x; u.h[1] = (__bf16)f0.y;
  u.h[2] = (__bf16)f0.z; u.h[3] = (__bf16)f0.w;
  u.h[4] = (__bf16)f1.x; u.h[5] = (__bf16)f1.y;
  u.h[6] = (__bf16)f1.z; u.h[7] = (__bf16)f1.w;
  *(uint4*)(dst + i) = u.q;
}

// Issue the async copies for one 128x64 bf16 tile: 256 threads, 64B each.
static __device__ __forceinline__ void issue_tile_async(const __bf16* __restrict__ src,
                                                        int row0, int ld, int kk,
                                                        __bf16* dst) {
  const int tid  = threadIdx.x;
  const int row  = tid >> 1;
  const int colb = (tid & 1) * 32;
  const __bf16* g = src + (size_t)(row0 + row) * ld + kk + colb;
  __bf16* l = dst + row * LDT2 + colb;
#pragma unroll
  for (int i = 0; i < 4; ++i) async_copy_b128(g + i * 8, l + i * 8);
}

// ---------------------------------------------------------------------------
// Shared 128x128 GEMM: C = A @ W^T (+bias), A[M,K] bf16, W[N,K] bf16.
// Double-buffered LDS tiles filled with global_load_async_to_lds_b128,
// K-step 64 (2 WMMA k-chunks), 8 waves = 4(M) x 2(N), 16 WMMAs per step.
// EPI=0: scatter into Q[B,H,T,D], K[B,H,T,D], V^T[B,H,D,T] (bf16)
// EPI=1: fp32 out[M,N]
// ---------------------------------------------------------------------------
template<int EPI>
__global__ __launch_bounds__(256) void gemm_bf16_wmma(
    const __bf16* __restrict__ A, const __bf16* __restrict__ W,
    const float* __restrict__ bias, const int Kdim,
    __bf16* __restrict__ Qb, __bf16* __restrict__ Kb, __bf16* __restrict__ Vt,
    float* __restrict__ out) {
  __shared__ __bf16 As[2][128 * LDT2];   // 2 x 18KB
  __shared__ __bf16 Ws[2][128 * LDT2];
  const int bn = blockIdx.x, bm = blockIdx.y;
  const int tid = threadIdx.x, lane = tid & 31, wave = tid >> 5;
  const int wm = wave & 3, wn = wave >> 2;
  AccU acc[2][4];
#pragma unroll
  for (int i = 0; i < 2; ++i)
#pragma unroll
    for (int j = 0; j < 4; ++j) { v8f z = {}; acc[i][j].v = z; }

  const int NT = Kdim / 64;
  issue_tile_async(A, bm * 128, Kdim, 0, As[0]);
  issue_tile_async(W, bn * 128, Kdim, 0, Ws[0]);

  for (int it = 0; it < NT; ++it) {
    const int cur = it & 1;
    if (it + 1 < NT) {     // prefetch next tile into the other buffer
      issue_tile_async(A, bm * 128, Kdim, (it + 1) * 64, As[cur ^ 1]);
      issue_tile_async(W, bn * 128, Kdim, (it + 1) * 64, Ws[cur ^ 1]);
      WAIT_ASYNC(8);       // tile `it` landed; next tile's 8 ops stay in flight
    } else {
      WAIT_ASYNC(0);
    }
    __syncthreads();       // cross-wave visibility of the async fills
#pragma unroll
    for (int kc = 0; kc < 2; ++kc) {
      v16bf a0 = load_a_frag(As[cur] + kc * 32, wm * 32,      LDT2, lane);
      v16bf a1 = load_a_frag(As[cur] + kc * 32, wm * 32 + 16, LDT2, lane);
#pragma unroll
      for (int ni = 0; ni < 4; ++ni) {
        v16bf bf = load_b_frag(Ws[cur] + kc * 32, wn * 64 + ni * 16, LDT2, lane);
        acc[0][ni].v = wmma_bf16(a0, bf, acc[0][ni].v);
        acc[1][ni].v = wmma_bf16(a1, bf, acc[1][ni].v);
      }
    }
    __syncthreads();       // buffer consumed; safe to refill two iters later
  }

  const int half = lane >> 4, ln = lane & 15;
#pragma unroll
  for (int mi = 0; mi < 2; ++mi) {
#pragma unroll
    for (int ni = 0; ni < 4; ++ni) {
      const int nf = bn * 128 + wn * 64 + ni * 16 + ln;
      const float bv = bias[nf];
      if constexpr (EPI == 0) {
        const int sec = nf / Cc;           // 0=q 1=k 2=v
        const int rem = nf - sec * Cc;
        const int h = rem >> 7, d = rem & (Dd - 1);
#pragma unroll
        for (int r = 0; r < 8; ++r) {
          const int m = bm * 128 + wm * 32 + mi * 16 + half * 8 + r;
          const int b = m >> 11, t = m & (Tt - 1);
          const __bf16 o = (__bf16)(acc[mi][ni].f[r] + bv);
          if (sec == 2) {
            Vt[(((size_t)(b * Hh + h)) * Dd + d) * Tt + t] = o;
          } else {
            const size_t idx = (((size_t)(b * Hh + h)) * Tt + t) * Dd + d;
            if (sec == 0) Qb[idx] = o; else Kb[idx] = o;
          }
        }
      } else {
#pragma unroll
        for (int r = 0; r < 8; ++r) {
          const int m = bm * 128 + wm * 32 + mi * 16 + half * 8 + r;
          out[(size_t)m * Cc + nf] = acc[mi][ni].f[r] + bv;
        }
      }
    }
  }
}

// ---------------------------------------------------------------------------
// Causal flash attention. One wave = 16 query rows; key blocks of 32; online
// softmax in fp32 with wave32 shfl reductions; P bridged C-layout -> A-layout
// through a per-wave LDS buffer. K/V fragments load straight from global
// (entire bf16 K/V = 64MB, resident in MI455X's 192MB L2).
// ---------------------------------------------------------------------------
__global__ __launch_bounds__(256) void flash_attn_wmma(
    const __bf16* __restrict__ Qb, const __bf16* __restrict__ Kb,
    const __bf16* __restrict__ Vt, const int* __restrict__ amask,
    __bf16* __restrict__ y) {
  __shared__ __bf16 Pbuf[8 * 16 * 32];    // per-wave 16x32 P tile
  const int tid = threadIdx.x, lane = tid & 31, wave = tid >> 5;
  const int half = lane >> 4, ln = lane & 15;
  const int bh = blockIdx.y, b = bh / Hh, h = bh - b * Hh;
  const int q0 = blockIdx.x * 128 + wave * 16;

  const __bf16* Q  = Qb + (size_t)bh * Tt * Dd;
  const __bf16* Kp = Kb + (size_t)bh * Tt * Dd;
  const __bf16* Vp = Vt + (size_t)bh * Dd * Tt;
  __bf16* Pw = Pbuf + wave * (16 * 32);

  v16bf qf[4];              // Q fragments for all of D=128, kept in registers
#pragma unroll
  for (int c = 0; c < 4; ++c) qf[c] = load_a_frag(Q + c * 32, q0, Dd, lane);

  AccU o[8];
#pragma unroll
  for (int i = 0; i < 8; ++i) { v8f z = {}; o[i].v = z; }
  float mrow[8], lrow[8];
#pragma unroll
  for (int r = 0; r < 8; ++r) { mrow[r] = -INFINITY; lrow[r] = 0.f; }

  const float scale = 0.088388347648318447f;  // 1/sqrt(128)
  const float LOG2E = 1.4426950408889634f;
  const int smax = q0 + 15;

  for (int s0 = 0; s0 <= smax; s0 += 32) {
    AccU sacc[2];
    { v8f z = {}; sacc[0].v = z; sacc[1].v = z; }
#pragma unroll
    for (int tt = 0; tt < 2; ++tt) {
      const int krow = s0 + tt * 16 + ln;       // < T (32-aligned tiles)
      const __bf16* kb = Kp + (size_t)krow * Dd + half * 16;
#pragma unroll
      for (int c = 0; c < 4; ++c)
        sacc[tt].v = wmma_bf16(qf[c], load_b_row(kb + c * 32), sacc[tt].v);
    }

    float sv[2][8];         // scale + causal/pad mask in C/D fragment layout
#pragma unroll
    for (int tt = 0; tt < 2; ++tt) {
      const int s = s0 + tt * 16 + ln;
      const bool colok = (amask[b * Tt + s] != 0);
#pragma unroll
      for (int r = 0; r < 8; ++r) {
        const int q = q0 + half * 8 + r;
        sv[tt][r] = (colok && s <= q) ? sacc[tt].f[r] * scale : -INFINITY;
      }
    }

    float corr[8];
#pragma unroll
    for (int r = 0; r < 8; ++r) {
      float mx   = red_max16(fmaxf(sv[0][r], sv[1][r]));
      float mnew = fmaxf(mrow[r], mx);
      corr[r] = exp2f((mrow[r] - mnew) * LOG2E);   // 0 on first block
      mrow[r] = mnew;
    }

    float rs[8];
#pragma unroll
    for (int r = 0; r < 8; ++r) rs[r] = 0.f;
#pragma unroll
    for (int tt = 0; tt < 2; ++tt)
#pragma unroll
      for (int r = 0; r < 8; ++r) {
        float p = exp2f((sv[tt][r] - mrow[r]) * LOG2E);
        rs[r] += p;
        Pw[(half * 8 + r) * 32 + tt * 16 + ln] = (__bf16)p;
      }
#pragma unroll
    for (int r = 0; r < 8; ++r) lrow[r] = lrow[r] * corr[r] + red_sum16(rs[r]);
#pragma unroll
    for (int dt = 0; dt < 8; ++dt)
#pragma unroll
      for (int r = 0; r < 8; ++r) o[dt].f[r] *= corr[r];

    // Per-wave DS ops are in-order in HW; just stop compiler reordering.
    __builtin_amdgcn_wave_barrier();
    v16bf pf = load_a_frag(Pw, 0, 32, lane);
#pragma unroll
    for (int dt = 0; dt < 8; ++dt) {
      const __bf16* vb = Vp + (size_t)(dt * 16 + ln) * Tt + s0 + half * 16;
      o[dt].v = wmma_bf16(pf, load_b_row(vb), o[dt].v);
    }
    __builtin_amdgcn_wave_barrier();
  }

  // y[B,T,H,D] == [M, C] bf16 feeds the output projection
#pragma unroll
  for (int dt = 0; dt < 8; ++dt) {
    const int d = dt * 16 + ln;
#pragma unroll
    for (int r = 0; r < 8; ++r) {
      const int q = q0 + half * 8 + r;
      y[((size_t)(b * Tt + q)) * Cc + h * Dd + d] = (__bf16)(o[dt].f[r] / lrow[r]);
    }
  }
}

// ---------------------------------------------------------------------------
extern "C" void kernel_launch(void* const* d_in, const int* in_sizes, int n_in,
                              void* d_out, int out_size, void* d_ws, size_t ws_size,
                              hipStream_t stream) {
  (void)in_sizes; (void)n_in; (void)out_size;
  const float* x     = (const float*)d_in[0];
  const int*   amask = (const int*)d_in[1];
  const float* wqkv  = (const float*)d_in[2];
  const float* bqkv  = (const float*)d_in[3];
  const float* wproj = (const float*)d_in[4];
  const float* bproj = (const float*)d_in[5];
  float* out = (float*)d_out;

  const size_t MB = 1024 * 1024;
  char* ws = (char*)d_ws;
  __bf16* xb  = (__bf16*)(ws);             // 32 MiB  x (bf16)
  __bf16* wqb = (__bf16*)(ws + 32 * MB);   // 24 MiB  w_qkv (bf16)
  __bf16* wpb = (__bf16*)(ws + 56 * MB);   //  8 MiB  w_proj (bf16)
  __bf16* Qb  = (__bf16*)(ws + 64 * MB);   // 32 MiB  Q [B,H,T,D]
  __bf16* Kb  = (__bf16*)(ws + 96 * MB);   // 32 MiB  K [B,H,T,D]
  __bf16* Vt  = (__bf16*)(ws + 128 * MB);  // 32 MiB  V^T [B,H,D,T]
  __bf16* yb  = (__bf16*)(ws + 160 * MB);  // 32 MiB  attn out [M,C]
  if (ws_size < 192 * MB) return;

  dim3 blk(256, 1, 1);
  const int nx = Mm * Cc, nq = N_QKV * Cc, np = Cc * Cc;
  cvt_f32_to_bf16<<<nx / (8 * 256), blk, 0, stream>>>(x, xb, nx);
  cvt_f32_to_bf16<<<nq / (8 * 256), blk, 0, stream>>>(wqkv, wqb, nq);
  cvt_f32_to_bf16<<<np / (8 * 256), blk, 0, stream>>>(wproj, wpb, np);

  gemm_bf16_wmma<0><<<dim3(N_QKV / 128, Mm / 128), blk, 0, stream>>>(
      xb, wqb, bqkv, Cc, Qb, Kb, Vt, nullptr);
  flash_attn_wmma<<<dim3(Tt / 128, Bsz * Hh), blk, 0, stream>>>(Qb, Kb, Vt,
                                                                amask, yb);
  gemm_bf16_wmma<1><<<dim3(Cc / 128, Mm / 128), blk, 0, stream>>>(
      yb, wpb, bproj, Cc, nullptr, nullptr, nullptr, out);
}
